// AdaptiveGraphAttentionModule_48747878810249
// MI455X (gfx1250) — matrix operand order repaired and checked by reading, not compiled, not run on hardware
//
#include <hip/hip_runtime.h>

// ---------------------------------------------------------------------------
// AdaptiveGraphAttention fused pipeline for MI455X (gfx1250, wave32, WMMA).
//   B=16, N=1024, H_DIM=128, HEADS=8, HEAD_DIM=16, scale=0.25
// attn_reg_loss = 1e-4 * mean(|softmax|) = 1e-4 / N  (rows sum to 1) -> constant.
//
// adj (the 512 MB logical / 32 MB unique operand) is streamed with CDNA5
// GLOBAL_LOAD_ASYNC_TO_LDS_B128 into a per-wave ping-pong LDS buffer,
// synchronized with s_wait_asynccnt (ASYNCcnt completes in order).
// ---------------------------------------------------------------------------

typedef __attribute__((ext_vector_type(16))) _Float16 v16h;
typedef __attribute__((ext_vector_type(8)))  _Float16 v8h;
typedef __attribute__((ext_vector_type(8)))  float    v8f;
typedef __attribute__((ext_vector_type(4)))  float    v4f;

namespace {
constexpr int Bc    = 16;
constexpr int Nc    = 1024;
constexpr int HD    = 128;   // model dim
constexpr int HEADS = 8;
constexpr int DH    = 16;    // head dim

__device__ inline v8f wmma_f16(const v16h& a, const v16h& b, const v8f& c) {
  // D = A(16x32 f16) * B(32x16 f16) + C(16x16 f32)
  return __builtin_amdgcn_wmma_f32_16x16x32_f16(false, a, false, b, (short)0, c,
                                                false, false);
}
} // namespace

// ---------------------------------------------------------------------------
// Kernel 1: Q/K/V projections.  y[.,o] = sum_k x[.,k] * W[o,k] + bias[o]
// One wave computes one 16x16 output tile (one head's d-range) with 4 WMMAs.
// Q,K stored f16 as [b][h][n][d]; V stored f16 transposed as [b][h][d][n].
// ---------------------------------------------------------------------------
__global__ __launch_bounds__(256) void qkv_proj_kernel(
    const float* __restrict__ x,
    const float* __restrict__ Wq, const float* __restrict__ bq,
    const float* __restrict__ Wk, const float* __restrict__ bk,
    const float* __restrict__ Wv, const float* __restrict__ bv,
    _Float16* __restrict__ Qh, _Float16* __restrict__ Kh,
    _Float16* __restrict__ Vt)
{
  const int lane  = threadIdx.x & 31;
  const int wv    = threadIdx.x >> 5;
  const int wflat = blockIdx.x * 8 + wv;     // 0 .. 1024*24-1
  const int mt    = wflat / 24;              // row tile 0..1023
  const int rem   = wflat % 24;
  const int mat   = rem >> 3;                // 0=Q 1=K 2=V
  const int ot    = rem & 7;                 // output col tile == head index
  const int m0    = mt * 16;                 // global row (over B*N)
  const int b     = m0 >> 10;                // / N
  const int n0    = m0 & (Nc - 1);

  const float* W    = (mat == 0) ? Wq : (mat == 1) ? Wk : Wv;
  const float* bias = (mat == 0) ? bq : (mat == 1) ? bk : bv;

  const int col   = lane & 15;
  const int halfw = lane >> 4;               // which 16-lane half
  const int bselA = halfw ? 8 : 0;           // A-frag K sub-offset
  const int bselB = halfw ? 8 : 0;           // B-frag K sub-offset
  const int r     = m0 + col;                // A row handled by this lane
  const int o     = ot * 16 + col;           // output column

  v8f c; for (int i = 0; i < 8; ++i) c[i] = 0.f;

  for (int kc = 0; kc < HD; kc += 32) {
    // A-frag: lane holds row r, K = {kc+bselA..+7} and {kc+16+bselA..+7}
    const float* ap = x + (size_t)r * HD + kc;
    v4f a0 = *(const v4f*)(ap + bselA);
    v4f a1 = *(const v4f*)(ap + bselA + 4);
    v4f a2 = *(const v4f*)(ap + 16 + bselA);
    v4f a3 = *(const v4f*)(ap + 16 + bselA + 4);
    v16h a;
    for (int j = 0; j < 4; ++j) {
      a[j]      = (_Float16)a0[j];
      a[4 + j]  = (_Float16)a1[j];
      a[8 + j]  = (_Float16)a2[j];
      a[12 + j] = (_Float16)a3[j];
    }
    // B-frag (W^T): lane holds column o, K = {kc+bselB..+7} and {kc+16+bselB..+7}
    const float* bp = W + (size_t)o * HD + kc + bselB;
    v4f b0 = *(const v4f*)(bp);
    v4f b1 = *(const v4f*)(bp + 4);
    v4f b2 = *(const v4f*)(bp + 16);
    v4f b3 = *(const v4f*)(bp + 20);
    v16h bf;
    for (int j = 0; j < 4; ++j) {
      bf[j]      = (_Float16)b0[j];
      bf[4 + j]  = (_Float16)b1[j];
      bf[8 + j]  = (_Float16)b2[j];
      bf[12 + j] = (_Float16)b3[j];
    }
    c = wmma_f16(a, bf, c);
  }

  const float bb = bias[o];
  const int h = ot, d = col;
  if (mat < 2) {
    _Float16* dst = (mat == 0) ? Qh : Kh;
    for (int i = 0; i < 8; ++i) {
      const int row = n0 + i + halfw * 8;
      dst[(((size_t)(b * HEADS + h)) * Nc + row) * DH + d] =
          (_Float16)(c[i] + bb);
    }
  } else {
    for (int i = 0; i < 8; ++i) {
      const int row = n0 + i + halfw * 8;
      Vt[(((size_t)(b * HEADS + h)) * DH + d) * Nc + row] =
          (_Float16)(c[i] + bb);
    }
  }
}

// ---------------------------------------------------------------------------
// Kernel 2: flash attention.  One wave owns one 16-query tile of one (b,h),
// streams 1024 keys in chunks of 32: 2 score WMMAs (K padded 16->32) +
// online softmax (f32, shfl_xor row reductions) + 1 P@V WMMA (full K=32).
// adj tiles (16x32 f32 = 2 KB/chunk) are prefetched one chunk ahead with
// GLOBAL_LOAD_ASYNC_TO_LDS_B128 into per-wave double buffers (ASYNCcnt).
// Result written f16 as [b*N + n][h*16 + d] for the output projection.
// ---------------------------------------------------------------------------
__global__ __launch_bounds__(256) void attn_kernel(
    const _Float16* __restrict__ Qh, const _Float16* __restrict__ Kh,
    const _Float16* __restrict__ Vt, const float* __restrict__ adj,
    _Float16* __restrict__ Ao)
{
  __shared__ __align__(16) _Float16 ldsP[8][16 * 32];      // per-wave P tile
  __shared__ __align__(16) float    adjBuf[8][2][16 * 32]; // per-wave adj pingpong

  const int lane   = threadIdx.x & 31;
  const int wv     = threadIdx.x >> 5;
  const int bh     = blockIdx.x >> 3;       // b*HEADS + h
  const int tchunk = blockIdx.x & 7;
  const int t      = tchunk * 8 + wv;       // query tile 0..63
  const int b      = bh >> 3;
  const int h      = bh & 7;                // scalar (blockIdx-only)
  const int col    = lane & 15;
  const int halfw  = lane >> 4;
  const int n0     = t * 16;
  const float scale = 0.25f;                // 1/sqrt(16)

  const _Float16* Qbh = Qh + (size_t)bh * Nc * DH;
  const _Float16* Kbh = Kh + (size_t)bh * Nc * DH;
  const _Float16* Vbh = Vt + (size_t)bh * DH * Nc;

  // --- async adj prefetch setup -------------------------------------------
  // SGPR base: adj + h*N*N (uniform, pure blockIdx expression).
  const float* adj_base = adj + (size_t)h * Nc * Nc;
  // Per-lane 32-bit byte offset of this lane's 16B slice of tile row group 0:
  //   row = n0 + (lane>>3), col bytes = (lane&7)*16; instruction i adds 4 rows.
  const unsigned voff0 =
      (unsigned)(((n0 + (lane >> 3)) * Nc + (lane & 7) * 4) * 4);
  // Raw LDS byte offsets of the two buffers (flat-address low 32 bits ==
  // LDS offset per the gfx1250 aperture rules), plus this lane's 16B slice.
  const unsigned ldsA =
      (unsigned)(size_t)(&adjBuf[wv][0][0]) + (unsigned)(lane * 16);
  const unsigned ldsB =
      (unsigned)(size_t)(&adjBuf[wv][1][0]) + (unsigned)(lane * 16);

  auto issue_adj = [&](int k0, int bufsel) {
    const unsigned ldsb = bufsel ? ldsB : ldsA;
    const unsigned gob  = voff0 + (unsigned)(k0 * 4);
#pragma unroll
    for (int i = 0; i < 4; ++i) {
      unsigned ldsaddr = ldsb + (unsigned)(i * 512);        // 4 rows * 128B
      unsigned goff    = gob + (unsigned)(i * 4 * Nc * 4);  // +4 tensor rows
      asm volatile("global_load_async_to_lds_b128 %0, %1, %2"
                   :: "v"(ldsaddr), "v"(goff), "s"(adj_base)
                   : "memory");
    }
  };
  // ------------------------------------------------------------------------

  // Q A-frag (fixed for whole key loop): row = col, lanes 0-15 d = 0..15, lanes 16-31 pad 0.
  v16h qa;
  {
    if (halfw == 0) {
      qa = *(const v16h*)(Qbh + ((size_t)(n0 + col)) * DH);
    } else {
      for (int j = 0; j < 16; ++j) qa[j] = (_Float16)0.f;
    }
  }

  v8f acc; float Mr[8], Lr[8];
  for (int i = 0; i < 8; ++i) { acc[i] = 0.f; Mr[i] = -1e30f; Lr[i] = 0.f; }

  issue_adj(/*k0=*/0, /*bufsel=*/0);   // prologue prefetch

  for (int ci = 0; ci < Nc / 32; ++ci) {
    const int k0 = ci * 32;

    // Prefetch next adj tile, then wait for the current one (in-order
    // completion: <=4 outstanding means the previous 4 B128s have landed).
    if (ci + 1 < Nc / 32) {
      issue_adj(k0 + 32, (ci + 1) & 1);
      asm volatile("s_wait_asynccnt 4" ::: "memory");
    } else {
      asm volatile("s_wait_asynccnt 0" ::: "memory");
    }
    const float* abuf = &adjBuf[wv][ci & 1][0];

    // K^T B-frags: lanes 0-15 carry d=0..15 (contiguous K row), lanes 16-31 pad 0.
    v16h kb0, kb1;
    if (halfw == 0) {
      kb0 = *(const v16h*)(Kbh + ((size_t)(k0 + col)) * DH);
      kb1 = *(const v16h*)(Kbh + ((size_t)(k0 + 16 + col)) * DH);
    } else {
      for (int j = 0; j < 16; ++j) { kb0[j] = (_Float16)0.f; kb1[j] = (_Float16)0.f; }
    }
    v8f z8; for (int i = 0; i < 8; ++i) z8[i] = 0.f;
    v8f s0 = wmma_f16(qa, kb0, z8);   // scores, keys k0..k0+15
    v8f s1 = wmma_f16(qa, kb1, z8);   // scores, keys k0+16..k0+31

    // scale + adj bias (from prefetched LDS tile, row-major 16x32)
    for (int i = 0; i < 8; ++i) {
      const int rloc = i + halfw * 8;
      s0[i] = s0[i] * scale + abuf[rloc * 32 + col];
      s1[i] = s1[i] * scale + abuf[rloc * 32 + 16 + col];
    }

    // online softmax update (row stats live redundantly in each 16-lane half)
    float p0[8], p1[8];
    for (int i = 0; i < 8; ++i) {
      float m = fmaxf(s0[i], s1[i]);
      m = fmaxf(m, __shfl_xor(m, 1, 32));
      m = fmaxf(m, __shfl_xor(m, 2, 32));
      m = fmaxf(m, __shfl_xor(m, 4, 32));
      m = fmaxf(m, __shfl_xor(m, 8, 32));
      const float newM  = fmaxf(Mr[i], m);
      const float alpha = __expf(Mr[i] - newM);
      const float e0 = __expf(s0[i] - newM);
      const float e1 = __expf(s1[i] - newM);
      float ls = e0 + e1;
      ls += __shfl_xor(ls, 1, 32);
      ls += __shfl_xor(ls, 2, 32);
      ls += __shfl_xor(ls, 4, 32);
      ls += __shfl_xor(ls, 8, 32);
      Lr[i]  = Lr[i] * alpha + ls;
      acc[i] = acc[i] * alpha;
      Mr[i]  = newM;
      p0[i] = e0; p1[i] = e1;
    }

    // P: C-layout -> LDS (row-major 16x32 f16) -> A-layout re-swizzle
    for (int i = 0; i < 8; ++i) {
      const int row = i + halfw * 8;
      ldsP[wv][row * 32 + col]      = (_Float16)p0[i];
      ldsP[wv][row * 32 + 16 + col] = (_Float16)p1[i];
    }
    v16h pa;
    {
      const int bsel = halfw ? 16 : 0;
      v8h x0 = *(const v8h*)(&ldsP[wv][col * 32 + bsel]);
      v8h x1 = *(const v8h*)(&ldsP[wv][col * 32 + bsel + 8]);
      for (int j = 0; j < 8; ++j) { pa[j] = x0[j]; pa[8 + j] = x1[j]; }
    }

    // V B-frag: lane holds column d=col, keys k0 + (halfw?16:0) + 0..15
    // (contiguous because V is stored d-major).
    v16h vb = *(const v16h*)(Vbh + (size_t)col * Nc + k0 + (halfw ? 16 : 0));
    acc = wmma_f16(pa, vb, acc);
  }

  // finalize (divide by softmax denom) and store f16 for output projection
  for (int i = 0; i < 8; ++i) {
    const int row = n0 + i + halfw * 8;
    const float oval = acc[i] / Lr[i];
    Ao[((size_t)(b * Nc + row)) * HD + h * DH + col] = (_Float16)oval;
  }
}

// ---------------------------------------------------------------------------
// Kernel 3: output projection out = Ao @ Wo^T + bo (f32 out), plus the
// analytically-constant regularization loss in the last output slot.
// ---------------------------------------------------------------------------
__global__ __launch_bounds__(256) void out_proj_kernel(
    const _Float16* __restrict__ A, const float* __restrict__ Wo,
    const float* __restrict__ bo, float* __restrict__ out)
{
  const int lane  = threadIdx.x & 31;
  const int wv    = threadIdx.x >> 5;
  const int wflat = blockIdx.x * 8 + wv;   // 0..8191
  const int mt    = wflat >> 3;
  const int ot    = wflat & 7;
  const int m0    = mt * 16;
  const int col   = lane & 15;
  const int halfw = lane >> 4;
  const int o     = ot * 16 + col;

  v8f c; for (int i = 0; i < 8; ++i) c[i] = 0.f;
  for (int kc = 0; kc < HD; kc += 32) {
    const int bsel = halfw ? 16 : 0;
    v8h a0 = *(const v8h*)(A + (size_t)(m0 + col) * HD + kc + bsel);
    v8h a1 = *(const v8h*)(A + (size_t)(m0 + col) * HD + kc + bsel + 8);
    v16h a;
    for (int j = 0; j < 8; ++j) { a[j] = a0[j]; a[8 + j] = a1[j]; }
    const float* bp = Wo + (size_t)o * HD + kc + (halfw ? 16 : 0);
    v4f b0 = *(const v4f*)(bp);
    v4f b1 = *(const v4f*)(bp + 4);
    v4f b2 = *(const v4f*)(bp + 8);
    v4f b3 = *(const v4f*)(bp + 12);
    v16h bf;
    for (int j = 0; j < 4; ++j) {
      bf[j]      = (_Float16)b0[j];
      bf[4 + j]  = (_Float16)b1[j];
      bf[8 + j]  = (_Float16)b2[j];
      bf[12 + j] = (_Float16)b3[j];
    }
    c = wmma_f16(a, bf, c);
  }
  const float bb = bo[o];
  for (int i = 0; i < 8; ++i) {
    const int row = m0 + i + halfw * 8;
    out[(size_t)row * HD + o] = c[i] + bb;
  }
  // attn_reg_loss = 1e-4 * mean(|softmax|) = 1e-4 / N  (exact)
  if (blockIdx.x == 0 && threadIdx.x == 0)
    out[(size_t)Bc * Nc * HD] = 1e-4f / 1024.0f;
}

// ---------------------------------------------------------------------------
extern "C" void kernel_launch(void* const* d_in, const int* in_sizes, int n_in,
                              void* d_out, int out_size, void* d_ws,
                              size_t ws_size, hipStream_t stream) {
  (void)in_sizes; (void)n_in; (void)out_size; (void)ws_size;
  const float* x   = (const float*)d_in[0];
  const float* Wq  = (const float*)d_in[1];
  const float* bq  = (const float*)d_in[2];
  const float* Wk  = (const float*)d_in[3];
  const float* bk  = (const float*)d_in[4];
  const float* Wv  = (const float*)d_in[5];
  const float* bv  = (const float*)d_in[6];
  const float* Wo  = (const float*)d_in[7];
  const float* bo  = (const float*)d_in[8];
  const float* adj = (const float*)d_in[9];
  float* out = (float*)d_out;

  char* ws = (char*)d_ws;
  const size_t MB4 = (size_t)4 * 1024 * 1024;
  _Float16* Qh = (_Float16*)(ws);            // [B][H][N][16] f16, 4 MB
  _Float16* Kh = (_Float16*)(ws + MB4);      // [B][H][N][16] f16, 4 MB
  _Float16* Vt = (_Float16*)(ws + 2 * MB4);  // [B][H][16][N] f16, 4 MB
  _Float16* Ao = (_Float16*)(ws + 3 * MB4);  // [B*N][128]    f16, 4 MB

  // 1024 row-tiles x (3 matrices x 8 head-tiles) = 24576 wave-tiles / 8 waves
  qkv_proj_kernel<<<3072, 256, 0, stream>>>(x, Wq, bq, Wk, bk, Wv, bv,
                                            Qh, Kh, Vt);
  // (B*H=128) x 8 query-tile chunks = 1024 blocks, 1 query tile per wave
  attn_kernel<<<1024, 256, 0, stream>>>(Qh, Kh, Vt, adj, Ao);
  // 1024 row-tiles x 8 col-tiles = 8192 wave-tiles / 8 waves
  out_proj_kernel<<<1024, 256, 0, stream>>>(Ao, Wo, bo, out);
}
